// Mamba_input_77610059039108
// MI455X (gfx1250) — matrix operand order repaired and checked by reading, not compiled, not run on hardware
//
#include <hip/hip_runtime.h>
#include <hip/hip_bf16.h>

typedef __attribute__((ext_vector_type(16))) _Float16 v16h;
typedef __attribute__((ext_vector_type(8)))  _Float16 v8h;
typedef __attribute__((ext_vector_type(4)))  _Float16 v4h;
typedef __attribute__((ext_vector_type(2)))  _Float16 v2h;
typedef __attribute__((ext_vector_type(8)))  float    v8f;
typedef __attribute__((ext_vector_type(4)))  float    v4f;
typedef __attribute__((ext_vector_type(2)))  float    v2f;
typedef __attribute__((ext_vector_type(4)))  int      v4i;

#define D_MODEL  1024
#define D_STATE  16
#define D_CONV   4
#define D_INNER  2048
#define DT_RANK  64
#define B_SZ     4
#define LSEQ     2048
#define MROWS    (B_SZ * LSEQ)      // 8192

#define BK       32
#define LPAD     8
#define KSTRIDE  (BK + LPAD)        // 40 halves = 80B rows (16B aligned)

// ---------------------------------------------------------------------------
// CDNA5 async global->LDS staging (ASYNCcnt), with safe fallback.
// Builtin signature (from hipcc diagnostic): (v4i as1*, v4i as3*, imm, imm).
// ---------------------------------------------------------------------------
#if defined(__HIP_DEVICE_COMPILE__) && defined(__has_builtin)
#  if __has_builtin(__builtin_amdgcn_global_load_async_to_lds_b128)
#    define HAVE_ASYNC_LDS 1
#  endif
#endif
#ifndef HAVE_ASYNC_LDS
#  define HAVE_ASYNC_LDS 0
#endif

// Copy 16 bytes (8 halves) from global to LDS.
__device__ __forceinline__ void stage16(const _Float16* g, _Float16* l) {
#if HAVE_ASYNC_LDS
  __builtin_amdgcn_global_load_async_to_lds_b128(
      (__attribute__((address_space(1))) v4i*)(void*)g,
      (__attribute__((address_space(3))) v4i*)(void*)l, 0, 0);
#else
  *(v8h*)l = *(const v8h*)g;   // global->VGPR->LDS fallback
#endif
}

__device__ __forceinline__ void wait_async0() {
#if HAVE_ASYNC_LDS
#  if __has_builtin(__builtin_amdgcn_s_wait_asynccnt)
  __builtin_amdgcn_s_wait_asynccnt(0);
#  else
  asm volatile("s_wait_asynccnt 0x0" ::: "memory");
#  endif
#endif
}

// Build a 16-half WMMA fragment from two 16B-aligned LDS chunks.
__device__ __forceinline__ v16h frag_from(const _Float16* p0, const _Float16* p1) {
  v8h lo = *(const v8h*)p0;
  v8h hi = *(const v8h*)p1;
  v16h r;
#pragma unroll
  for (int i = 0; i < 8; ++i) { r[i] = lo[i]; r[i + 8] = hi[i]; }
  return r;
}

// ---------------------------------------------------------------------------
// Prep: f32 -> f16 convert (vectorized x4), and A = -exp(A_log)
// ---------------------------------------------------------------------------
__global__ void cvt_f32_f16_kernel(const float* __restrict__ src,
                                   _Float16* __restrict__ dst, int n4) {
  int i = blockIdx.x * blockDim.x + threadIdx.x;
  if (i < n4) {
    float4 v = ((const float4*)src)[i];
    v4h h;
    h[0] = (_Float16)v.x; h[1] = (_Float16)v.y;
    h[2] = (_Float16)v.z; h[3] = (_Float16)v.w;
    *(v4h*)(dst + (size_t)i * 4) = h;
  }
}

__global__ void neg_exp_kernel(const float* __restrict__ src,
                               float* __restrict__ dst, int n) {
  int i = blockIdx.x * blockDim.x + threadIdx.x;
  if (i < n) dst[i] = -expf(src[i]);
}

// ---------------------------------------------------------------------------
// GEMM1: C(8192 x 2048) = H_f16(8192 x 1024) * W1_f16(2048 x 1024)^T
// Double-buffered async LDS staging; one barrier per K-step.
// Block tile 128x128x32, 8 waves (2 M x 4 N), each wave 64x32 = 4x2 WMMA tiles.
// Output transposed (via LDS) as f16 tensor Xraw laid out (b, d, l).
// ---------------------------------------------------------------------------
#define G1_BUF 20480   // bytes per stage buffer (As 10240 + Bs 10240)

__global__ __launch_bounds__(256) void gemm1_kernel(
    const _Float16* __restrict__ Ag,   // (8192, 1024)
    const _Float16* __restrict__ Bg,   // (2048, 1024)
    _Float16* __restrict__ Xraw)       // (4, 2048, 2048) as (b,d,l)
{
  __shared__ alignas(16) char smem[40960];  // 2 stage buffers; Cs(34816) overlays
  _Float16 (*Cs)[136] = reinterpret_cast<_Float16(*)[136]>(smem);

  const int tid   = threadIdx.x;
  const int lane  = tid & 31;
  const int wave  = tid >> 5;
  const int wM    = (wave & 1) * 64;
  const int wN    = (wave >> 1) * 32;
  const int lrow  = lane & 15;
  const int lhalf = lane >> 4;          // 0/1
  const int aK0   = lhalf * 8;          // A frag: halves 0..7 -> K aK0..+7, 8..15 -> K aK0+16..+23
  const int bK0   = lhalf * 8;          // B frag: halves 0..7 -> K bK0..+7, 8..15 -> K bK0+16..+23

  const int mBlock = blockIdx.y * 128;
  const int nBlock = blockIdx.x * 128;
  const int ldRow  = tid >> 1;          // 0..127
  const int ldCol  = (tid & 1) * 16;    // 0 or 16

  const _Float16* srcA = Ag + (size_t)(mBlock + ldRow) * D_MODEL + ldCol;
  const _Float16* srcB = Bg + (size_t)(nBlock + ldRow) * D_MODEL + ldCol;
  const int ldsOff = ldRow * KSTRIDE + ldCol;

  auto stage_tile = [&](int k0, int buf) {
    _Float16* a = (_Float16*)(smem + buf * G1_BUF) + ldsOff;
    _Float16* b = (_Float16*)(smem + buf * G1_BUF + 10240) + ldsOff;
    stage16(srcA + k0,     a);
    stage16(srcA + k0 + 8, a + 8);
    stage16(srcB + k0,     b);
    stage16(srcB + k0 + 8, b + 8);
  };

  v8f acc[4][2] = {};
  stage_tile(0, 0);
  const int KT = D_MODEL / BK;          // 32
  for (int kt = 0; kt < KT; ++kt) {
    wait_async0();
    __syncthreads();                    // tile kt visible; buf^1 reads from kt-1 done
    if (kt + 1 < KT) stage_tile((kt + 1) * BK, (kt + 1) & 1);

    const _Float16 (*As)[KSTRIDE] =
        (const _Float16(*)[KSTRIDE])(smem + (kt & 1) * G1_BUF);
    const _Float16 (*Bs)[KSTRIDE] =
        (const _Float16(*)[KSTRIDE])(smem + (kt & 1) * G1_BUF + 10240);

    v16h bfrag[2];
#pragma unroll
    for (int j = 0; j < 2; ++j) {
      const _Float16* p = &Bs[wN + j * 16 + lrow][bK0];
      bfrag[j] = frag_from(p, p + 16);
    }
#pragma unroll
    for (int i = 0; i < 4; ++i) {
      const _Float16* pa = &As[wM + i * 16 + lrow][aK0];
      v16h afrag = frag_from(pa, pa + 16);
#pragma unroll
      for (int j = 0; j < 2; ++j)
        acc[i][j] = __builtin_amdgcn_wmma_f32_16x16x32_f16(
            false, afrag, false, bfrag[j], (short)0, acc[i][j], false, false);
    }
  }

  __syncthreads();
  // Transpose accumulators into Cs[n][m] (f16). C layout: VGPR r -> M=r(+8 hi lanes), lane&15 -> N.
#pragma unroll
  for (int i = 0; i < 4; ++i)
#pragma unroll
    for (int j = 0; j < 2; ++j) {
      const int n  = wN + j * 16 + lrow;
      const int mB = wM + i * 16 + lhalf * 8;
#pragma unroll
      for (int r = 0; r < 8; r += 2) {
        v2h pr;
        pr[0] = (_Float16)acc[i][j][r];
        pr[1] = (_Float16)acc[i][j][r + 1];
        *(v2h*)&Cs[n][mB + r] = pr;
      }
    }
  __syncthreads();

  const int b  = mBlock >> 11;
  const int l0 = mBlock & (LSEQ - 1);
#pragma unroll
  for (int it = 0; it < 8; ++it) {
    int c   = tid + it * 256;           // 2048 chunks of 8 halves
    int n   = c >> 4;
    int col = (c & 15) * 8;
    v8h v = *(const v8h*)&Cs[n][col];
    *(v8h*)(Xraw + (size_t)(b * D_INNER + nBlock + n) * LSEQ + l0 + col) = v;
  }
}

// ---------------------------------------------------------------------------
// Causal depthwise conv(4) + bias + SiLU.
// Reads Xraw f16 (b,d,l); writes x fp32 (b,d,l) to d_out and f16 (b,l,d) to ws.
// ---------------------------------------------------------------------------
#define CV_D 32
#define CV_L 64
__global__ __launch_bounds__(256) void conv_silu_kernel(
    const _Float16* __restrict__ Xraw,
    const float* __restrict__ Wc,      // (2048, 1, 4)
    const float* __restrict__ Bc,      // (2048)
    float* __restrict__ Xout,          // (b, d, l) fp32
    _Float16* __restrict__ XT)         // (b, l, d) f16
{
  __shared__ _Float16 tile[CV_D][CV_L + 4];           // halo of 4 on the left
  __shared__ alignas(16) _Float16 xt[CV_L][CV_D + 8]; // 40-half rows, 16B aligned

  const int blk = blockIdx.x;
  const int lt  = blk & 31;
  const int dt  = (blk >> 5) & 63;
  const int b   = blk >> 11;
  const int l0  = lt * CV_L;
  const int d0  = dt * CV_D;
  const int tid = threadIdx.x;

  for (int idx = tid; idx < CV_D * (CV_L + 4); idx += 256) {
    int d  = idx / (CV_L + 4);
    int lc = idx % (CV_L + 4);
    int lg = l0 - 4 + lc;
    _Float16 v = (_Float16)0.f;
    if (lg >= 0) v = Xraw[(size_t)(b * D_INNER + d0 + d) * LSEQ + lg];
    tile[d][lc] = v;
  }
  __syncthreads();

  const int dLoc  = tid >> 3;           // 0..31
  const int lBase = (tid & 7) * 8;      // 0..56
  const int dg    = d0 + dLoc;
  const float w0 = Wc[dg * 4 + 0], w1 = Wc[dg * 4 + 1];
  const float w2 = Wc[dg * 4 + 2], w3 = Wc[dg * 4 + 3];
  const float bias = Bc[dg];
  float* outp = Xout + (size_t)(b * D_INNER + dg) * LSEQ + l0 + lBase;
#pragma unroll
  for (int q = 0; q < 8; ++q) {
    int lc = lBase + q;
    // output l = l0+lc uses inputs l-3..l -> tile cols lc+1..lc+4
    float y = bias
            + w0 * (float)tile[dLoc][lc + 1]
            + w1 * (float)tile[dLoc][lc + 2]
            + w2 * (float)tile[dLoc][lc + 3]
            + w3 * (float)tile[dLoc][lc + 4];
    y = y / (1.f + __expf(-y));         // SiLU
    outp[q] = y;
    xt[lc][dLoc] = (_Float16)y;
  }
  __syncthreads();
  {
    int l    = tid >> 2;
    int dcol = (tid & 3) * 8;
    v8h v = *(const v8h*)&xt[l][dcol];
    *(v8h*)(XT + (size_t)(b * LSEQ + l0 + l) * D_INNER + d0 + dcol) = v;
  }
}

// ---------------------------------------------------------------------------
// GEMM2: x_dbl(8192 x 96) = XT_f16(8192 x 2048) * x_proj_f16(96 x 2048)^T
// 8 waves (4 M x 2 N); wave tile 32x48 = 2x3 WMMA tiles.
// Epilogue: n<64 -> dt_part f16; 64..79 -> B_mat; 80..95 -> C_mat.
// ---------------------------------------------------------------------------
#define G2_BUF 17920   // As 10240 + Bs 7680

__global__ __launch_bounds__(256) void gemm2_kernel(
    const _Float16* __restrict__ Ag,   // (8192, 2048)
    const _Float16* __restrict__ Bg,   // (96, 2048)
    _Float16* __restrict__ DTp,        // (8192, 64) f16
    float* __restrict__ outB,          // (4, 16, 2048)
    float* __restrict__ outC)          // (4, 16, 2048)
{
  __shared__ alignas(16) char smem[35840];  // 2 stage buffers

  const int tid   = threadIdx.x;
  const int lane  = tid & 31;
  const int wave  = tid >> 5;
  const int wM    = (wave & 3) * 32;
  const int wN    = (wave >> 2) * 48;
  const int lrow  = lane & 15;
  const int lhalf = lane >> 4;
  const int aK0   = lhalf * 8;
  const int bK0   = lhalf * 8;

  const int mBlock = blockIdx.y * 128;
  const int ldRow  = tid >> 1;
  const int ldCol  = (tid & 1) * 16;

  const _Float16* srcA = Ag + (size_t)(mBlock + ldRow) * D_INNER + ldCol;
  const _Float16* srcB = Bg + (size_t)ldRow * D_INNER + ldCol;  // valid for ldRow<96
  const int ldsOff = ldRow * KSTRIDE + ldCol;

  auto stage_tile = [&](int k0, int buf) {
    _Float16* a = (_Float16*)(smem + buf * G2_BUF) + ldsOff;
    stage16(srcA + k0,     a);
    stage16(srcA + k0 + 8, a + 8);
    if (ldRow < 96) {                 // waves 6,7 fully skip (wave-aligned)
      _Float16* b = (_Float16*)(smem + buf * G2_BUF + 10240) + ldsOff;
      stage16(srcB + k0,     b);
      stage16(srcB + k0 + 8, b + 8);
    }
  };

  v8f acc[2][3] = {};
  stage_tile(0, 0);
  const int KT = D_INNER / BK;          // 64
  for (int kt = 0; kt < KT; ++kt) {
    wait_async0();
    __syncthreads();
    if (kt + 1 < KT) stage_tile((kt + 1) * BK, (kt + 1) & 1);

    const _Float16 (*As)[KSTRIDE] =
        (const _Float16(*)[KSTRIDE])(smem + (kt & 1) * G2_BUF);
    const _Float16 (*Bs)[KSTRIDE] =
        (const _Float16(*)[KSTRIDE])(smem + (kt & 1) * G2_BUF + 10240);

    v16h bfrag[3];
#pragma unroll
    for (int j = 0; j < 3; ++j) {
      const _Float16* p = &Bs[wN + j * 16 + lrow][bK0];
      bfrag[j] = frag_from(p, p + 16);
    }
#pragma unroll
    for (int i = 0; i < 2; ++i) {
      const _Float16* pa = &As[wM + i * 16 + lrow][aK0];
      v16h afrag = frag_from(pa, pa + 16);
#pragma unroll
      for (int j = 0; j < 3; ++j)
        acc[i][j] = __builtin_amdgcn_wmma_f32_16x16x32_f16(
            false, afrag, false, bfrag[j], (short)0, acc[i][j], false, false);
    }
  }

  const int b  = mBlock >> 11;
  const int l0 = mBlock & (LSEQ - 1);
#pragma unroll
  for (int i = 0; i < 2; ++i)
#pragma unroll
    for (int j = 0; j < 3; ++j) {
      const int n  = wN + j * 16 + lrow;            // each 16-wide tile is branch-uniform
      const int mB = wM + i * 16 + lhalf * 8;
#pragma unroll
      for (int r = 0; r < 8; ++r) {
        const int m = mB + r;
        const int l = l0 + m;
        float v = acc[i][j][r];
        if (n < DT_RANK) {
          DTp[(size_t)(mBlock + m) * DT_RANK + n] = (_Float16)v;
        } else if (n < DT_RANK + D_STATE) {
          outB[((size_t)b * D_STATE + (n - DT_RANK)) * LSEQ + l] = v;
        } else {
          outC[((size_t)b * D_STATE + (n - DT_RANK - D_STATE)) * LSEQ + l] = v;
        }
      }
    }
}

// ---------------------------------------------------------------------------
// GEMM3: dt(8192 x 2048) = dtp_f16(8192 x 64) * dt_proj_f16(2048 x 64)^T
// Output fp32, transposed (b, d, l) via two half-tile LDS passes.
// ---------------------------------------------------------------------------
#define G3_BUF 20480

__global__ __launch_bounds__(256) void gemm3_kernel(
    const _Float16* __restrict__ Ag,   // (8192, 64)
    const _Float16* __restrict__ Bg,   // (2048, 64)
    float* __restrict__ DtOut)         // (4, 2048, 2048) as (b,d,l)
{
  __shared__ alignas(16) char smem[40960];  // 2 stage buffers; Cs(64*132*4=33792) overlays
  float (*Cs)[132] = reinterpret_cast<float(*)[132]>(smem);

  const int tid   = threadIdx.x;
  const int lane  = tid & 31;
  const int wave  = tid >> 5;
  const int wM    = (wave & 1) * 64;
  const int wN    = (wave >> 1) * 32;
  const int lrow  = lane & 15;
  const int lhalf = lane >> 4;
  const int aK0   = lhalf * 8;
  const int bK0   = lhalf * 8;

  const int mBlock = blockIdx.y * 128;
  const int nBlock = blockIdx.x * 128;
  const int ldRow  = tid >> 1;
  const int ldCol  = (tid & 1) * 16;

  const _Float16* srcA = Ag + (size_t)(mBlock + ldRow) * DT_RANK + ldCol;
  const _Float16* srcB = Bg + (size_t)(nBlock + ldRow) * DT_RANK + ldCol;
  const int ldsOff = ldRow * KSTRIDE + ldCol;

  auto stage_tile = [&](int k0, int buf) {
    _Float16* a = (_Float16*)(smem + buf * G3_BUF) + ldsOff;
    _Float16* b = (_Float16*)(smem + buf * G3_BUF + 10240) + ldsOff;
    stage16(srcA + k0,     a);
    stage16(srcA + k0 + 8, a + 8);
    stage16(srcB + k0,     b);
    stage16(srcB + k0 + 8, b + 8);
  };

  v8f acc[4][2] = {};
  stage_tile(0, 0);
  const int KT = DT_RANK / BK;          // 2
  for (int kt = 0; kt < KT; ++kt) {
    wait_async0();
    __syncthreads();
    if (kt + 1 < KT) stage_tile((kt + 1) * BK, (kt + 1) & 1);

    const _Float16 (*As)[KSTRIDE] =
        (const _Float16(*)[KSTRIDE])(smem + (kt & 1) * G3_BUF);
    const _Float16 (*Bs)[KSTRIDE] =
        (const _Float16(*)[KSTRIDE])(smem + (kt & 1) * G3_BUF + 10240);

    v16h bfrag[2];
#pragma unroll
    for (int j = 0; j < 2; ++j) {
      const _Float16* p = &Bs[wN + j * 16 + lrow][bK0];
      bfrag[j] = frag_from(p, p + 16);
    }
#pragma unroll
    for (int i = 0; i < 4; ++i) {
      const _Float16* pa = &As[wM + i * 16 + lrow][aK0];
      v16h afrag = frag_from(pa, pa + 16);
#pragma unroll
      for (int j = 0; j < 2; ++j)
        acc[i][j] = __builtin_amdgcn_wmma_f32_16x16x32_f16(
            false, afrag, false, bfrag[j], (short)0, acc[i][j], false, false);
    }
  }

  const int b  = mBlock >> 11;
  const int l0 = mBlock & (LSEQ - 1);
  // Two passes: waves with wN<64 write n 0..63; waves with wN>=64 write n 64..127.
  for (int half = 0; half < 2; ++half) {
    __syncthreads();
    if ((wN >= 64) == (half == 1)) {
#pragma unroll
      for (int i = 0; i < 4; ++i)
#pragma unroll
        for (int j = 0; j < 2; ++j) {
          const int n  = wN + j * 16 + lrow - half * 64;
          const int mB = wM + i * 16 + lhalf * 8;
#pragma unroll
          for (int r = 0; r < 8; r += 2) {
            v2f p;
            p[0] = acc[i][j][r];
            p[1] = acc[i][j][r + 1];
            *(v2f*)&Cs[n][mB + r] = p;
          }
        }
    }
    __syncthreads();
#pragma unroll
    for (int it = 0; it < 8; ++it) {
      int c   = tid + it * 256;          // 2048 chunks of 4 floats (64 rows x 32)
      int n   = c >> 5;
      int col = (c & 31) * 4;
      v4f v = *(const v4f*)&Cs[n][col];
      *(v4f*)(DtOut + (size_t)(b * D_INNER + nBlock + half * 64 + n) * LSEQ +
              l0 + col) = v;
    }
  }
}

// ---------------------------------------------------------------------------
// Launch
// ---------------------------------------------------------------------------
extern "C" void kernel_launch(void* const* d_in, const int* in_sizes, int n_in,
                              void* d_out, int out_size, void* d_ws, size_t ws_size,
                              hipStream_t stream) {
  const float* H    = (const float*)d_in[0];  // (4, 2048, 1024)
  const float* W1   = (const float*)d_in[1];  // (4096, 1024) — only first 2048 rows needed
  const float* Wc   = (const float*)d_in[2];  // (2048, 1, 4)
  const float* Bc   = (const float*)d_in[3];  // (2048)
  const float* XP   = (const float*)d_in[4];  // (96, 2048)
  const float* DTPw = (const float*)d_in[5];  // (2048, 64)
  const float* Alog = (const float*)d_in[6];  // (2048, 16)
  float* out = (float*)d_out;

  // workspace layout (bytes); total ~89.8 MB
  char* ws = (char*)d_ws;
  _Float16* Hf    = (_Float16*)(ws);               // 16,777,216 B
  _Float16* W1f   = (_Float16*)(ws + 16777216);    //  4,194,304 B
  _Float16* XPf   = (_Float16*)(ws + 20971520);    //    393,216 B
  _Float16* DTPf  = (_Float16*)(ws + 21364736);    //    262,144 B
  _Float16* Xraw  = (_Float16*)(ws + 21626880);    // 33,554,432 B (b,d,l)
  _Float16* XT    = (_Float16*)(ws + 55181312);    // 33,554,432 B (b,l,d)
  _Float16* DTp16 = (_Float16*)(ws + 88735744);    //  1,048,576 B (8192,64)

  // output layout (floats, tuple order: dt, A, B_mat, C_mat, x)
  float* dt_out = out;
  float* A_out  = out + 16777216;
  float* B_out  = out + 16809984;
  float* C_out  = out + 16941056;
  float* x_out  = out + 17072128;

  // Stage 0: precision converts + A = -exp(A_log)
  cvt_f32_f16_kernel<<<(8388608 / 4 + 255) / 256, 256, 0, stream>>>(H, Hf, 8388608 / 4);
  cvt_f32_f16_kernel<<<(2097152 / 4 + 255) / 256, 256, 0, stream>>>(W1, W1f, 2097152 / 4);
  cvt_f32_f16_kernel<<<(196608 / 4 + 255) / 256, 256, 0, stream>>>(XP, XPf, 196608 / 4);
  cvt_f32_f16_kernel<<<(131072 / 4 + 255) / 256, 256, 0, stream>>>(DTPw, DTPf, 131072 / 4);
  neg_exp_kernel<<<(32768 + 255) / 256, 256, 0, stream>>>(Alog, A_out, 32768);

  // Stage 1: in_proj (x half only) -> Xraw (b,d,l) f16
  gemm1_kernel<<<dim3(D_INNER / 128, MROWS / 128), 256, 0, stream>>>(Hf, W1f, Xraw);

  // Stage 2: conv + SiLU -> x (fp32 out) and XT (b,l,d) f16
  conv_silu_kernel<<<B_SZ * (D_INNER / CV_D) * (LSEQ / CV_L), 256, 0, stream>>>(
      Xraw, Wc, Bc, x_out, XT);

  // Stage 3: x_proj -> dt_part (f16), B_mat, C_mat
  gemm2_kernel<<<dim3(1, MROWS / 128), 256, 0, stream>>>(XT, XPf, DTp16, B_out, C_out);

  // Stage 4: dt_proj -> dt (b,d,l) fp32
  gemm3_kernel<<<dim3(D_INNER / 128, MROWS / 128), 256, 0, stream>>>(DTp16, DTPf, dt_out);
}
